// BlockSparseTopKLinear_89962384982177
// MI455X (gfx1250) — matrix operand order repaired and checked by reading, not compiled, not run on hardware
//
#include <hip/hip_runtime.h>

#define IN_F  4096
#define OUT_F 11008
#define BM    32
#define BK    64
#define NBM   4      // 128 / 32 row-blocks
#define NBK   64     // 4096 / 64 k-blocks
#define KEEP  32     // top-k kept per row-block
#define NTILE 128    // N columns per workgroup
#define KTILE 128    // K rows staged per LDS tile (2 k-blocks)
#define LDSK  136    // padded LDS row stride in halves (272B: 16B aligned, bank-friendly)

typedef _Float16 h8    __attribute__((ext_vector_type(8)));
typedef _Float16 v16h  __attribute__((ext_vector_type(16)));
typedef float    v8f   __attribute__((ext_vector_type(8)));
typedef unsigned u32x4 __attribute__((ext_vector_type(4)));

// ---------------------------------------------------------------------------
// Kernel 1: per row-block activation score + top-k (keep=32) block selection.
// One block per row-block. Deterministic fixed-order reduction.
// ws layout: ws[0..3] = active count per rb; ws[4 + rb*64 + i] = active kb ids.
// ---------------------------------------------------------------------------
__global__ __launch_bounds__(256) void topk_mask_kernel(
    const _Float16* __restrict__ x, int* __restrict__ ws) {
  __shared__ float sPart[4][NBK];
  __shared__ float sAct[NBK];
  __shared__ int   sMask[NBK];

  const int rb  = blockIdx.x;
  const int t   = threadIdx.x;
  const int kb  = t & 63;   // k-block this thread scores
  const int grp = t >> 6;   // row group 0..3 (8 rows each)

  float s = 0.f;
  const _Float16* xb = x + (size_t)(rb * BM + grp * 8) * IN_F + kb * BK;
  for (int r = 0; r < 8; ++r) {
    const h8* p = (const h8*)(xb + (size_t)r * IN_F);
#pragma unroll
    for (int c = 0; c < 8; ++c) {
      h8 v = p[c];
#pragma unroll
      for (int e = 0; e < 8; ++e) s += __builtin_fabsf((float)v[e]);
    }
  }
  sPart[grp][kb] = s;
  __syncthreads();

  if (t < NBK)  // fixed-order combine of the 4 partials -> deterministic
    sAct[t] = sPart[0][t] + sPart[1][t] + sPart[2][t] + sPart[3][t];
  __syncthreads();

  if (t < NBK) {
    // act_i >= kth_largest(KEEP)  <=>  #{j : act_j > act_i} < KEEP  (tie-safe)
    const float a = sAct[t];
    int cnt = 0;
    for (int j = 0; j < NBK; ++j) cnt += (sAct[j] > a) ? 1 : 0;
    sMask[t] = (cnt < KEEP) ? 1 : 0;
  }
  __syncthreads();

  if (t == 0) {
    int n = 0;
    for (int j = 0; j < NBK; ++j)
      if (sMask[j]) ws[4 + rb * NBK + n++] = j;
    ws[rb] = n;
  }
}

// ---------------------------------------------------------------------------
// Kernel 2: block-sparse GEMM with v_wmma_f32_16x16x32_f16.
// Grid: (OUT_F/128, 4 row-blocks). 256 threads = 8 wave32.
// Workgroup output tile: 32(M) x 128(N). Wave w owns M-tile (w&1) and
// N-tiles (w>>1) and (w>>1)+4.
// Each LDS tile stages TWO active K-blocks (128 K rows) transposed to [n][k];
// double-buffered -> 8 WMMAs per wave per barrier. Odd tail: second half of
// tile is zero-filled (contributes exactly 0), keeping control flow uniform.
// ---------------------------------------------------------------------------
__global__ __launch_bounds__(256) void bs_gemm_kernel(
    const _Float16* __restrict__ x, const _Float16* __restrict__ w,
    const _Float16* __restrict__ bias, const int* __restrict__ ws,
    float* __restrict__ out) {
  __shared__ _Float16 Bt[2][NTILE * LDSK];  // [buf][n][k], ~34KB each

  const int n0    = blockIdx.x * NTILE;
  const int rb    = blockIdx.y;
  const int t     = threadIdx.x;
  const int lane  = t & 31;
  const int wv    = t >> 5;      // wave 0..7
  const int mt    = wv & 1;      // M tile 0..1
  const int ntb   = wv >> 1;     // N tile base 0..3 (second tile = +4)
  const int lhalf = lane >> 4;   // 0..1
  const int lmod  = lane & 15;

  const int  nActive = ws[rb];
  const int* list    = ws + 4 + rb * NBK;
  const int  ntiles  = (nActive + 1) >> 1;

  // Stage two 64(K) x 128(N) weight blocks into one LDS tile, transposed to
  // [n][k]. Per thread: 4 pair-segments; each = 2 coalesced b128 global reads
  // + 8 packed b32 LDS stores (one v_perm_b32 per dword). kb1 < 0 => zeros.
  auto fill2 = [&](int buf, int kb0, int kb1) {
    _Float16* dst = &Bt[buf][0];
#pragma unroll
    for (int pp = 0; pp < 4; ++pp) {
      const int p  = t + pp * 256;     // pair id 0..1023
      const int kk = (p >> 4) * 2;     // even local K row 0..126
      const int ns = (p & 15) * 8;     // N segment start 0..120
      unsigned pk[8];
      if (pp < 2 || kb1 >= 0) {        // uniform branch
        const int    kbq = (pp < 2) ? kb0 : kb1;
        const size_t g   = (size_t)(kbq * BK + (kk & 63)) * OUT_F + n0 + ns;
        u32x4 u0 = *(const u32x4*)(w + g);
        u32x4 u1 = *(const u32x4*)(w + g + OUT_F);
#pragma unroll
        for (int jj = 0; jj < 4; ++jj) {
#if __has_builtin(__builtin_amdgcn_perm)
          pk[2 * jj]     = __builtin_amdgcn_perm(u1[jj], u0[jj], 0x05040100u);
          pk[2 * jj + 1] = __builtin_amdgcn_perm(u1[jj], u0[jj], 0x07060302u);
#else
          pk[2 * jj]     = (u0[jj] & 0xffffu) | (u1[jj] << 16);
          pk[2 * jj + 1] = (u0[jj] >> 16) | (u1[jj] & 0xffff0000u);
#endif
        }
      } else {
#pragma unroll
        for (int jj = 0; jj < 8; ++jj) pk[jj] = 0u;
      }
#pragma unroll
      for (int e = 0; e < 8; ++e)
        *(unsigned*)(dst + (size_t)(ns + e) * LDSK + kk) = pk[e];
    }
  };

  v8f acc0 = {}, acc1 = {};

  if (ntiles > 0)
    fill2(0, list[0], (nActive > 1) ? list[1] : -1);
  __syncthreads();

  const int row = rb * BM + mt * 16 + lmod;  // A row M = lane%16 (both halves)

  for (int j = 0; j < ntiles; ++j) {
    const int cur = j & 1;
    if (j + 1 < ntiles) {
      const int i1 = 2 * (j + 1);
      fill2(cur ^ 1, list[i1], (i1 + 1 < nActive) ? list[i1 + 1] : -1);
    }

    // Prefetch tile j+2 toward L2 (256 threads x 128B covers the 32KB tile).
    if (j + 2 < ntiles) {
      const int i2  = 2 * (j + 2);
      const int pk0 = list[i2];
      const int pk1 = (i2 + 1 < nActive) ? list[i2 + 1] : pk0;
      const int kk  = t >> 1;
      const int kbq = (kk < 64) ? pk0 : pk1;
      __builtin_prefetch(
          w + (size_t)(kbq * BK + (kk & 63)) * OUT_F + n0 + (t & 1) * 64, 0, 0);
    }

    const int       kb0  = list[2 * j];
    const int       kb1e = (2 * j + 1 < nActive) ? list[2 * j + 1] : kb0;
    const _Float16* base = &Bt[cur][0];
#pragma unroll
    for (int ks = 0; ks < 4; ++ks) {  // four K=32 WMMA steps per 128-K tile
      const int kbq = (ks < 2) ? kb0 : kb1e;
      // A fragment straight from global (x is K-contiguous):
      // lane l: K run [16*lhalf .. +15] (contiguous, matches B fragment).
      const _Float16* ap =
          x + (size_t)row * IN_F + kbq * BK + (ks & 1) * 32 + lhalf * 16;
      h8 a0 = *(const h8*)ap;
      h8 a1 = *(const h8*)(ap + 8);
      v16h a;
#pragma unroll
      for (int e = 0; e < 8; ++e) { a[e] = a0[e]; a[8 + e] = a1[e]; }

      // B fragments from LDS: lane l -> col n = nt*16 + l%16,
      // local K run = ks*32 + 16*lhalf .. +15 (contiguous in the [n][k] row).
      const _Float16* bp0 =
          base + (size_t)(ntb * 16 + lmod) * LDSK + ks * 32 + lhalf * 16;
      const _Float16* bp1 = bp0 + (size_t)64 * LDSK;  // second N tile (+64)
      h8 b0l = *(const h8*)bp0;
      h8 b0h = *(const h8*)(bp0 + 8);
      h8 b1l = *(const h8*)bp1;
      h8 b1h = *(const h8*)(bp1 + 8);
      v16h b0, b1;
#pragma unroll
      for (int e = 0; e < 8; ++e) {
        b0[e] = b0l[e]; b0[8 + e] = b0h[e];
        b1[e] = b1l[e]; b1[8 + e] = b1h[e];
      }

      acc0 = __builtin_amdgcn_wmma_f32_16x16x32_f16(
          false, a, false, b0, (short)0, acc0, false, false);
      acc1 = __builtin_amdgcn_wmma_f32_16x16x32_f16(
          false, a, false, b1, (short)0, acc1, false, false);
    }
    __syncthreads();
  }

  // C/D layout: VGPR r -> M = r + 8*(lane/16), N = lane%16.
  const int   gcol0 = n0 + ntb * 16 + lmod;
  const int   gcol1 = gcol0 + 64;
  const float bv0   = (float)bias[gcol0];
  const float bv1   = (float)bias[gcol1];
  const int   grow0 = rb * BM + mt * 16 + 8 * lhalf;
#pragma unroll
  for (int r = 0; r < 8; ++r) {
    out[(size_t)(grow0 + r) * OUT_F + gcol0] = acc0[r] + bv0;
    out[(size_t)(grow0 + r) * OUT_F + gcol1] = acc1[r] + bv1;
  }
}

extern "C" void kernel_launch(void* const* d_in, const int* in_sizes, int n_in,
                              void* d_out, int out_size, void* d_ws, size_t ws_size,
                              hipStream_t stream) {
  const _Float16* x    = (const _Float16*)d_in[0];
  const _Float16* w    = (const _Float16*)d_in[1];
  const _Float16* bias = (const _Float16*)d_in[2];
  float*          out  = (float*)d_out;
  int*            ws   = (int*)d_ws;

  topk_mask_kernel<<<dim3(NBM), 256, 0, stream>>>(x, ws);
  bs_gemm_kernel<<<dim3(OUT_F / NTILE, NBM), 256, 0, stream>>>(x, w, bias, ws, out);
}
